// FlashAttention_32263794327916
// MI455X (gfx1250) — matrix operand order repaired and checked by reading, not compiled, not run on hardware
//
#include <hip/hip_runtime.h>
#include <hip/hip_bf16.h>

#define B_  2
#define S_  2048
#define D_  2048
#define H_  16
#define HD_ 128

typedef __attribute__((ext_vector_type(16))) __bf16       v16bf;
typedef __attribute__((ext_vector_type(8)))  float        v8f;
typedef __attribute__((ext_vector_type(4)))  unsigned int u32x4;
typedef __attribute__((ext_vector_type(4)))  int          i32x4;

union Frag16 { v16bf v; u32x4 q[2]; };

__device__ __forceinline__ unsigned short f32_bf16(float f) {
    unsigned int u = __float_as_uint(f);
    u += 0x7FFFu + ((u >> 16) & 1u);   // round-to-nearest-even
    return (unsigned short)(u >> 16);
}

__device__ __forceinline__ v8f zero8() {
    v8f z;
    for (int i = 0; i < 8; i++) z[i] = 0.0f;
    return z;
}

// Async global->LDS copy (CDNA5 ASYNCcnt path); sync fallback keeps compiles green.
__device__ __forceinline__ void async_b128(const void* g, void* l) {
#if __has_builtin(__builtin_amdgcn_global_load_async_to_lds_b128)
    __builtin_amdgcn_global_load_async_to_lds_b128(
        (__attribute__((address_space(1))) i32x4*)(void*)g,
        (__attribute__((address_space(3))) i32x4*)l, 0, 0);
#else
    *(u32x4*)l = *(const u32x4*)g;
#endif
}

__device__ __forceinline__ void wait_async0() {
#if __has_builtin(__builtin_amdgcn_s_wait_asynccnt)
    __builtin_amdgcn_s_wait_asynccnt(0);
#else
    asm volatile("s_wait_asynccnt 0" ::: "memory");
#endif
}

// ---------------------------------------------------------------------------
// fp32 -> bf16 elementwise convert
// ---------------------------------------------------------------------------
__global__ void cvt_f32_bf16(const float* __restrict__ in,
                             unsigned short* __restrict__ out, int n) {
    for (int i = blockIdx.x * blockDim.x + threadIdx.x; i < n;
         i += gridDim.x * blockDim.x)
        out[i] = f32_bf16(in[i]);
}

// ---------------------------------------------------------------------------
// C[M,N] = (A[M,K] * W[N,K]^T + bias) * scale   (A, W bf16; out bf16 or f32)
// 256 threads = 8 waves; 128x256 block tile; each wave 64x64 via 4x4 WMMA
// accumulators; K-tiles of 32, double-buffered async global->LDS staging.
// ---------------------------------------------------------------------------
__global__ __launch_bounds__(256) void gemm_bf16(
    const unsigned short* __restrict__ A, const unsigned short* __restrict__ W,
    const float* __restrict__ bias, unsigned short* __restrict__ outB,
    float* __restrict__ outF, int M, int N, int K, float scale)
{
    __shared__ __align__(16) unsigned short As[2][128 * 32];
    __shared__ __align__(16) unsigned short Bs[2][256 * 32];

    const int t    = threadIdx.x;
    const int lane = t & 31, w = t >> 5;
    const int wm = w & 1, wn = w >> 1;           // 2 wave-rows x 4 wave-cols
    const int m0 = blockIdx.y * 128, n0 = blockIdx.x * 256;
    const int lh = (lane < 16) ? 0 : 1;
    const int ln = lane & 15;

    v8f acc[4][4];
    for (int i = 0; i < 4; i++)
        for (int j = 0; j < 4; j++) acc[i][j] = zero8();

    auto prefetch = [&](int buf, int k0) {
        // A tile: 128 rows x 32 k = 512 x 16B chunks (2/thread)
        for (int i = 0; i < 2; i++) {
            int idx = t + i * 256;
            int row = idx >> 2, c = idx & 3;
            async_b128(&A[(size_t)(m0 + row) * K + k0 + c * 8],
                       &As[buf][row * 32 + c * 8]);
        }
        // B tile: 256 rows x 32 k = 1024 x 16B chunks (4/thread)
        for (int i = 0; i < 4; i++) {
            int idx = t + i * 256;
            int row = idx >> 2, c = idx & 3;
            async_b128(&W[(size_t)(n0 + row) * K + k0 + c * 8],
                       &Bs[buf][row * 32 + c * 8]);
        }
    };

    prefetch(0, 0);
    const int KT = K >> 5;
    for (int kt = 0; kt < KT; kt++) {
        const int cur = kt & 1;
        wait_async0();
        __syncthreads();
        if (kt + 1 < KT) prefetch(cur ^ 1, (kt + 1) * 32);

        Frag16 af[4];
        for (int mi = 0; mi < 4; mi++) {
            int m  = wm * 64 + mi * 16 + ln;
            int bs = m * 32 + lh * 8;
            af[mi].q[0] = *(const u32x4*)&As[cur][bs];
            af[mi].q[1] = *(const u32x4*)&As[cur][bs + 16];
        }
        for (int ni = 0; ni < 4; ni++) {
            Frag16 bf;
            int n  = wn * 64 + ni * 16 + ln;
            int bs = n * 32 + lh * 8;
            bf.q[0] = *(const u32x4*)&Bs[cur][bs];
            bf.q[1] = *(const u32x4*)&Bs[cur][bs + 16];
            for (int mi = 0; mi < 4; mi++)
                acc[mi][ni] = __builtin_amdgcn_wmma_f32_16x16x32_bf16(
                    false, af[mi].v, false, bf.v, (short)0, acc[mi][ni],
                    false, false);
        }
        __syncthreads();
    }

    for (int mi = 0; mi < 4; mi++)
        for (int ni = 0; ni < 4; ni++) {
            int   n  = n0 + wn * 64 + ni * 16 + ln;
            float bv = bias ? bias[n] : 0.0f;
            for (int r = 0; r < 8; r++) {
                int   m   = m0 + wm * 64 + mi * 16 + r + lh * 8;
                float val = (acc[mi][ni][r] + bv) * scale;
                if (outF) outF[(size_t)m * N + n] = val;
                else      outB[(size_t)m * N + n] = f32_bf16(val);
            }
        }
}

// ---------------------------------------------------------------------------
// Causal flash attention. grid = (S/128, H, B), block = 256 (8 waves).
// Each wave owns 16 query rows; key blocks of 64; Q pre-scaled by 1/sqrt(HD).
// Q/K/V laid out [B*S, D] with head h at column offset h*HD.
// ---------------------------------------------------------------------------
__global__ __launch_bounds__(256) void flash_attn(
    const unsigned short* __restrict__ Qb, const unsigned short* __restrict__ Kb,
    const unsigned short* __restrict__ Vb, unsigned short* __restrict__ O)
{
    __shared__ __align__(16) unsigned short smem[8192 * 3];
    unsigned short* Klds = smem;            // [64 keys][128 hd]
    unsigned short* Vt   = smem + 8192;     // [128 hd][64 keys]   (transposed)
    unsigned short* Plds = smem + 16384;    // 8 waves * [16 q][64 keys]

    const int t = threadIdx.x, lane = t & 31, w = t >> 5;
    const int lh = (lane < 16) ? 0 : 1, ln = lane & 15;
    const int q0 = blockIdx.x * 128;
    const int h  = blockIdx.y, b = blockIdx.z;
    const size_t base = ((size_t)b * S_) * D_ + (size_t)h * HD_;

    // stage Q tile asynchronously (reuses K/Vt smem), build per-wave A frags
    for (int i = 0; i < 8; i++) {
        int idx = t + i * 256;
        int row = idx >> 4, c = idx & 15;
        async_b128(&Qb[base + (size_t)(q0 + row) * D_ + c * 8],
                   &smem[row * 128 + c * 8]);
    }
    wait_async0();
    __syncthreads();
    Frag16 qf[4];
    {
        int qrow = w * 16 + ln;
        for (int ds = 0; ds < 4; ds++) {
            int bs = qrow * 128 + ds * 32 + lh * 8;
            qf[ds].q[0] = *(const u32x4*)&smem[bs];
            qf[ds].q[1] = *(const u32x4*)&smem[bs + 16];
        }
    }
    __syncthreads();

    v8f ov[8];
    for (int i = 0; i < 8; i++) ov[i] = zero8();
    float mrow[8], lrow[8];
    for (int r = 0; r < 8; r++) { mrow[r] = -1e30f; lrow[r] = 0.0f; }

    const int jEnd = q0 / 64 + 1;   // inclusive; causal upper bound for block
    for (int j = 0; j <= jEnd; j++) {
        const int kb0 = j * 64;
        // K block [64][128] via async global->LDS
        for (int i = 0; i < 4; i++) {
            int idx = t + i * 256;
            int kr = idx >> 4, c = idx & 15;
            async_b128(&Kb[base + (size_t)(kb0 + kr) * D_ + c * 8],
                       &Klds[kr * 128 + c * 8]);
        }
        // V block transposed: Vt[hd][key] (needs lane transpose -> sync path)
        for (int i = 0; i < 4; i++) {
            int idx = t + i * 256;
            int kr = idx >> 4, c = idx & 15;
            const unsigned short* src =
                &Vb[base + (size_t)(kb0 + kr) * D_ + c * 8];
            for (int e = 0; e < 8; e++) Vt[(c * 8 + e) * 64 + kr] = src[e];
        }
        wait_async0();
        __syncthreads();

        // scores: 16 q rows x 64 keys per wave (Q pre-scaled)
        v8f sc[4];
        for (int nf = 0; nf < 4; nf++) {
            sc[nf] = zero8();
            int key = nf * 16 + ln;
            for (int ds = 0; ds < 4; ds++) {
                Frag16 kf;
                int bs = key * 128 + ds * 32 + lh * 8;
                kf.q[0] = *(const u32x4*)&Klds[bs];
                kf.q[1] = *(const u32x4*)&Klds[bs + 16];
                sc[nf] = __builtin_amdgcn_wmma_f32_16x16x32_bf16(
                    false, qf[ds].v, false, kf.v, (short)0, sc[nf], false, false);
            }
        }
        // causal mask
        for (int nf = 0; nf < 4; nf++) {
            int kg = kb0 + nf * 16 + ln;
            for (int r = 0; r < 8; r++) {
                int qg = q0 + w * 16 + r + lh * 8;
                if (kg > qg) sc[nf][r] = -1e30f;
            }
        }
        // online softmax (row stats live per lane-half, 16-lane reductions)
        float corr[8];
        for (int r = 0; r < 8; r++) {
            float mx = sc[0][r];
            for (int nf = 1; nf < 4; nf++) mx = fmaxf(mx, sc[nf][r]);
            for (int off = 1; off < 16; off <<= 1)
                mx = fmaxf(mx, __shfl_xor(mx, off, 32));
            float mnew = fmaxf(mrow[r], mx);
            corr[r] = __expf(mrow[r] - mnew);
            float rs = 0.0f;
            for (int nf = 0; nf < 4; nf++) {
                float p = __expf(sc[nf][r] - mnew);
                sc[nf][r] = p;
                rs += p;
            }
            for (int off = 1; off < 16; off <<= 1) rs += __shfl_xor(rs, off, 32);
            lrow[r] = lrow[r] * corr[r] + rs;
            mrow[r] = mnew;
        }
        for (int df = 0; df < 8; df++)
            for (int r = 0; r < 8; r++) ov[df][r] *= corr[r];

        // reshape P (C-layout) -> A-layout via wave-private LDS scratch
        unsigned short* pw = Plds + w * 1024;
        for (int nf = 0; nf < 4; nf++) {
            int col = nf * 16 + ln;
            for (int r = 0; r < 8; r++)
                pw[(r + lh * 8) * 64 + col] = f32_bf16(sc[nf][r]);
        }
        asm volatile("s_wait_dscnt 0" ::: "memory");
        Frag16 pf[2];
        for (int ks = 0; ks < 2; ks++) {
            int bs = ln * 64 + ks * 32 + lh * 8;
            pf[ks].q[0] = *(const u32x4*)&pw[bs];
            pf[ks].q[1] = *(const u32x4*)&pw[bs + 16];
        }
        // O += P x V
        for (int df = 0; df < 8; df++) {
            int hd = df * 16 + ln;
            for (int ks = 0; ks < 2; ks++) {
                Frag16 vf;
                int bs = hd * 64 + ks * 32 + lh * 8;
                vf.q[0] = *(const u32x4*)&Vt[bs];
                vf.q[1] = *(const u32x4*)&Vt[bs + 16];
                ov[df] = __builtin_amdgcn_wmma_f32_16x16x32_bf16(
                    false, pf[ks].v, false, vf.v, (short)0, ov[df], false, false);
            }
        }
        __syncthreads();
    }

    // epilogue: O / l, bf16, layout [B*S, D] col h*HD+hd
    for (int r = 0; r < 8; r++) {
        float inv = 1.0f / lrow[r];
        int   qg  = q0 + w * 16 + r + lh * 8;
        size_t orow = ((size_t)b * S_ + qg) * D_ + (size_t)h * HD_;
        for (int df = 0; df < 8; df++) {
            int hd = df * 16 + ln;
            O[orow + hd] = f32_bf16(ov[df][r] * inv);
        }
    }
}

// ---------------------------------------------------------------------------
extern "C" void kernel_launch(void* const* d_in, const int* in_sizes, int n_in,
                              void* d_out, int out_size, void* d_ws,
                              size_t ws_size, hipStream_t stream) {
    (void)in_sizes; (void)n_in; (void)out_size; (void)ws_size;

    const float* x  = (const float*)d_in[0];
    const float* Wq = (const float*)d_in[1];
    const float* bq = (const float*)d_in[2];
    const float* Wk = (const float*)d_in[3];
    const float* bk = (const float*)d_in[4];
    const float* Wv = (const float*)d_in[5];
    const float* bv = (const float*)d_in[6];
    const float* Wo = (const float*)d_in[7];
    const float* bo = (const float*)d_in[8];
    float* out = (float*)d_out;

    const size_t nX = (size_t)B_ * S_ * D_;   // 8,388,608
    const size_t nW = (size_t)D_ * D_;        // 4,194,304

    char* ws = (char*)d_ws;
    size_t off = 0;
    unsigned short* xb  = (unsigned short*)(ws + off); off += nX * 2;
    unsigned short* wqb = (unsigned short*)(ws + off); off += nW * 2;
    unsigned short* wkb = (unsigned short*)(ws + off); off += nW * 2;
    unsigned short* wvb = (unsigned short*)(ws + off); off += nW * 2;
    unsigned short* wob = (unsigned short*)(ws + off); off += nW * 2;
    unsigned short* qb  = (unsigned short*)(ws + off); off += nX * 2;
    unsigned short* kbp = (unsigned short*)(ws + off); off += nX * 2;
    unsigned short* vbp = (unsigned short*)(ws + off); off += nX * 2;
    unsigned short* ab  = (unsigned short*)(ws + off); off += nX * 2;

    // 1) downconvert activations + weights to bf16
    cvt_f32_bf16<<<2048, 256, 0, stream>>>(x,  xb,  (int)nX);
    cvt_f32_bf16<<<1024, 256, 0, stream>>>(Wq, wqb, (int)nW);
    cvt_f32_bf16<<<1024, 256, 0, stream>>>(Wk, wkb, (int)nW);
    cvt_f32_bf16<<<1024, 256, 0, stream>>>(Wv, wvb, (int)nW);
    cvt_f32_bf16<<<1024, 256, 0, stream>>>(Wo, wob, (int)nW);

    // 2) Q/K/V projections (Q pre-scaled by 1/sqrt(HD))
    const int M = B_ * S_, N = D_, K = D_;
    dim3 gg(N / 256, M / 128);
    const float qscale = 0.088388347648318447f;  // 1/sqrt(128)
    gemm_bf16<<<gg, 256, 0, stream>>>(xb, wqb, bq, qb,  nullptr, M, N, K, qscale);
    gemm_bf16<<<gg, 256, 0, stream>>>(xb, wkb, bk, kbp, nullptr, M, N, K, 1.0f);
    gemm_bf16<<<gg, 256, 0, stream>>>(xb, wvb, bv, vbp, nullptr, M, N, K, 1.0f);

    // 3) causal flash attention
    dim3 ga(S_ / 128, H_, B_);
    flash_attn<<<ga, 256, 0, stream>>>(qb, kbp, vbp, ab);

    // 4) output projection -> fp32 d_out
    gemm_bf16<<<gg, 256, 0, stream>>>(ab, wob, bo, nullptr, out, M, N, K, 1.0f);
}
